// ScaledDotProdAttRelPosEncV1_52123723105014
// MI455X (gfx1250) — compile-verified, matrix operand order checked by reading, not run on hardware
//
#include <hip/hip_runtime.h>

// ---------------------------------------------------------------------------
// Transformer-XL relative-position attention for MI455X (gfx1250, wave32)
// All GEMMs on v_wmma_f32_16x16x32_f16 (f32 accumulate).
// Rel-shift identity: BD[t,s] = qv[t] . p[T-1-|t-s|]  (no extra masking).
// Round 2: one-shot f32->f16 pre-conversion so every GEMM streams pure f16
// fragments (half the L2 read bytes, no v_cvt in the WMMA inner loops).
// ---------------------------------------------------------------------------

typedef __attribute__((ext_vector_type(16))) _Float16 v16h;
typedef __attribute__((ext_vector_type(8)))  _Float16 v8h;
typedef __attribute__((ext_vector_type(8)))  float    v8f;
typedef __attribute__((ext_vector_type(4)))  float    f32x4;

#define WMMA_F16(a, b, c) \
  __builtin_amdgcn_wmma_f32_16x16x32_f16(false, (a), false, (b), (short)0, (c), false, false)

static constexpr int Bc   = 2;
static constexpr int Tc   = 1024;
static constexpr int Fc   = 1024;
static constexpr int Hc   = 16;
static constexpr int DKc  = 64;
static constexpr int DVc  = 64;
static constexpr int OUTc = 1024;
static constexpr int TS   = 32;     // t-rows per attention block

// ---- fragment loader --------------------------------------------------------
// 16-bit A/B fragment layout (ISA 7.12.2): lane l -> row r = l&15, half hi = l>>4.
// halves 0..7  = K[hi*8 .. hi*8+7], halves 8..15 = K[16+hi*8 .. 16+hi*8+7].
__device__ inline v16h frag_f16(const _Float16* base, int hi) {
  v8h lo = *(const v8h*)(base + hi * 8);
  v8h hh = *(const v8h*)(base + 16 + hi * 8);
  return __builtin_shufflevector(lo, hh, 0, 1, 2, 3, 4, 5, 6, 7,
                                         8, 9, 10, 11, 12, 13, 14, 15);
}

// ---------------------------------------------------------------------------
// One-shot f32 -> f16 conversion (vectorized: b128 load, b128 store per lane)
// ---------------------------------------------------------------------------
__global__ void __launch_bounds__(256)
cvt_kernel(const float* __restrict__ in, _Float16* __restrict__ out, int n) {
  const int i = (blockIdx.x * 256 + threadIdx.x) * 8;
  if (i + 8 <= n) {
    f32x4 a = *(const f32x4*)(in + i);
    f32x4 b = *(const f32x4*)(in + i + 4);
    v8h o;
#pragma unroll
    for (int k = 0; k < 4; ++k) {
      o[k]     = (_Float16)a[k];
      o[4 + k] = (_Float16)b[k];
    }
    *(v8h*)(out + i) = o;
  }
}

// ---------------------------------------------------------------------------
// Projection GEMM (all-f16 operands): C[m,n] = sum_f A[m,f]*W[n,f] + bias[n]
// mode 0: Q -> write (C+u_bias) and (C+v_bias) f16 at (b,h,t,d)
// mode 1: K -> write C f16 at (b,h,t,d)
// mode 2: V -> write C f16 transposed at (b,h,d,t)
// mode 3: P -> write C f16 at (h,t,d)    (b==0)
// Block tile 128x64, 8 waves x (2x2) 16x16 WMMA tiles, K-step 32.
// ---------------------------------------------------------------------------
__global__ void __launch_bounds__(256)
proj_kernel(const _Float16* __restrict__ A, const _Float16* __restrict__ W,
            const float* __restrict__ bias, int mode,
            _Float16* __restrict__ out0, _Float16* __restrict__ out1,
            const float* __restrict__ ub, const float* __restrict__ vb) {
  const int lane = threadIdx.x & 31;
  const int w    = threadIdx.x >> 5;
  const int r    = lane & 15;
  const int hi   = lane >> 4;
  const int m0   = blockIdx.x * 128 + (w & 3) * 32;
  const int n0   = blockIdx.y * 64 + (w >> 2) * 32;

  v8f acc[2][2] = {};

  const _Float16* arow0 = A + (size_t)(m0 + r) * Fc;
  const _Float16* arow1 = A + (size_t)(m0 + 16 + r) * Fc;
  const _Float16* wrow0 = W + (size_t)(n0 + r) * Fc;
  const _Float16* wrow1 = W + (size_t)(n0 + 16 + r) * Fc;

  for (int k0 = 0; k0 < Fc; k0 += 32) {
    __builtin_prefetch(arow0 + k0 + 256, 0, 1);
    __builtin_prefetch(wrow0 + k0 + 256, 0, 1);
    v16h a0 = frag_f16(arow0 + k0, hi);
    v16h a1 = frag_f16(arow1 + k0, hi);
    v16h b0 = frag_f16(wrow0 + k0, hi);
    v16h b1 = frag_f16(wrow1 + k0, hi);
    acc[0][0] = WMMA_F16(a0, b0, acc[0][0]);
    acc[0][1] = WMMA_F16(a0, b1, acc[0][1]);
    acc[1][0] = WMMA_F16(a1, b0, acc[1][0]);
    acc[1][1] = WMMA_F16(a1, b1, acc[1][1]);
  }

  // D layout: lane -> n = r; VGPR v -> m = v + 8*hi (ISA 7.12.2)
#pragma unroll
  for (int ti = 0; ti < 2; ++ti) {
#pragma unroll
    for (int tj = 0; tj < 2; ++tj) {
      const int n  = n0 + tj * 16 + r;
      const float bb = bias[n];
      const int h = n >> 6, d = n & 63;
#pragma unroll
      for (int v = 0; v < 8; ++v) {
        const int m = m0 + ti * 16 + v + 8 * hi;
        const int b = m >> 10;        // T == 1024
        const int t = m & 1023;
        const float c = acc[ti][tj][v] + bb;
        if (mode == 0) {
          const size_t idx = (((size_t)b * Hc + h) * Tc + t) * DKc + d;
          out0[idx] = (_Float16)(c + ub[n]);
          out1[idx] = (_Float16)(c + vb[n]);
        } else if (mode == 1) {
          out0[(((size_t)b * Hc + h) * Tc + t) * DKc + d] = (_Float16)c;
        } else if (mode == 2) {
          out0[(((size_t)b * Hc + h) * DVc + d) * Tc + t] = (_Float16)c;
        } else {
          out0[((size_t)h * Tc + t) * DKc + d] = (_Float16)c;
        }
      }
    }
  }
}

// ---------------------------------------------------------------------------
// Attention kernel: one block per (b, h, 32-row t-block). 256 threads / 8 waves.
// LDS: scores f32 [32][1024] | pb f16 [32][1024] (Braw, then probs) | red f32 [256]
// ---------------------------------------------------------------------------
__global__ void __launch_bounds__(256)
attn_kernel(const _Float16* __restrict__ qu, const _Float16* __restrict__ qv,
            const _Float16* __restrict__ kf, const _Float16* __restrict__ vT,
            const _Float16* __restrict__ pf, _Float16* __restrict__ x) {
  extern __shared__ char smem[];
  float*     scores = (float*)smem;                          // 131072 B
  _Float16*  pb     = (_Float16*)(smem + 131072);            //  65536 B
  float*     red    = (float*)(smem + 131072 + 65536);       //   1024 B

  const int bid = blockIdx.x;
  const int tb  = bid & 31;           // T/TS == 32
  const int h   = (bid >> 5) & 15;
  const int b   = bid >> 9;
  const int t0  = tb * TS;
  const int bh  = b * Hc + h;

  const int w     = threadIdx.x >> 5;
  const int lane  = threadIdx.x & 31;
  const int r     = lane & 15;
  const int hi    = lane >> 4;
  const int ttile = w & 1;

  // Preload q+u / q+v A-fragments for this wave's t-tile (K = 64 -> two frags)
  const _Float16* qurow = qu + (((size_t)bh) * Tc + t0 + ttile * 16 + r) * DKc;
  const _Float16* qvrow = qv + (((size_t)bh) * Tc + t0 + ttile * 16 + r) * DKc;
  const v16h aqu0 = frag_f16(qurow, hi), aqu1 = frag_f16(qurow + 32, hi);
  const v16h aqv0 = frag_f16(qvrow, hi), aqv1 = frag_f16(qvrow + 32, hi);

  // ---- Phase A: Braw[t, j] = qv[t] . p[j]  -> LDS f16 -----------------------
  for (int jt = (w >> 1); jt < 64; jt += 4) {
    const int j0 = jt * 16;
    const _Float16* prow = pf + ((size_t)h * Tc + j0 + r) * DKc;
    v16h b0 = frag_f16(prow, hi), b1 = frag_f16(prow + 32, hi);
    v8f acc = {};
    acc = WMMA_F16(aqv0, b0, acc);
    acc = WMMA_F16(aqv1, b1, acc);
#pragma unroll
    for (int v = 0; v < 8; ++v) {
      const int rrow = ttile * 16 + v + 8 * hi;
      pb[rrow * 1024 + j0 + r] = (_Float16)acc[v];
    }
  }
  __syncthreads();

  // ---- Phase B: scores = (qu.k + Braw[t, 1023-|t-s|]) / sqrt(DK) ------------
  for (int st = (w >> 1); st < 64; st += 4) {
    const int s0 = st * 16;
    const _Float16* krow = kf + (((size_t)bh) * Tc + s0 + r) * DKc;
    v16h b0 = frag_f16(krow, hi), b1 = frag_f16(krow + 32, hi);
    v8f acc = {};
    acc = WMMA_F16(aqu0, b0, acc);
    acc = WMMA_F16(aqu1, b1, acc);
    const int s = s0 + r;
#pragma unroll
    for (int v = 0; v < 8; ++v) {
      const int rrow = ttile * 16 + v + 8 * hi;
      const int t    = t0 + rrow;
      const int dd   = t - s;
      const int ad   = dd < 0 ? -dd : dd;
      const float bdv = (float)pb[rrow * 1024 + (1023 - ad)];
      scores[rrow * 1024 + s] = (acc[v] + bdv) * 0.125f;
    }
  }
  __syncthreads();

  // ---- Phase C: row softmax; probs -> pb (f16), overwriting Braw ------------
  {
    const int row = threadIdx.x >> 3;   // 32 rows x 8 threads
    const int ch  = threadIdx.x & 7;    // 128 cols each
    float* srow = scores + row * 1024 + ch * 128;
    float mx = -3.4e38f;
    for (int i = 0; i < 128; ++i) mx = fmaxf(mx, srow[i]);
    red[row * 8 + ch] = mx;
    __syncthreads();
    float rmx = red[row * 8];
#pragma unroll
    for (int i = 1; i < 8; ++i) rmx = fmaxf(rmx, red[row * 8 + i]);
    __syncthreads();
    float sm = 0.f;
    for (int i = 0; i < 128; ++i) {
      const float e = __expf(srow[i] - rmx);
      srow[i] = e;
      sm += e;
    }
    red[row * 8 + ch] = sm;
    __syncthreads();
    float rs = 0.f;
#pragma unroll
    for (int i = 0; i < 8; ++i) rs += red[row * 8 + i];
    const float inv = 1.f / rs;
    _Float16* prow = pb + row * 1024 + ch * 128;
    for (int i = 0; i < 128; ++i) prow[i] = (_Float16)(srow[i] * inv);
  }
  __syncthreads();

  // ---- Phase D: out[t,d] = sum_s P[t,s] * v[s,d]  (B-frags from vT rows) ----
  {
    const int dtile = w >> 1;
    const int dcol  = dtile * 16 + r;
    const _Float16* vrow = vT + (((size_t)bh) * DVc + dcol) * Tc;
    const _Float16* prow = pb + (size_t)(ttile * 16 + r) * 1024;
    v8f acc = {};
    for (int kk = 0; kk < Tc; kk += 32) {
      __builtin_prefetch(vrow + kk + 256, 0, 1);
      v16h a   = frag_f16(prow + kk, hi);
      v16h bfr = frag_f16(vrow + kk, hi);
      acc = WMMA_F16(a, bfr, acc);
    }
#pragma unroll
    for (int v = 0; v < 8; ++v) {
      const int t = t0 + ttile * 16 + v + 8 * hi;
      x[((size_t)b * Tc + t) * (Hc * DVc) + h * 64 + dcol] = (_Float16)acc[v];
    }
  }
}

// ---------------------------------------------------------------------------
// Output GEMM (all-f16 operands): y[m,n] = sum_k x[m,k]*Wo[n,k] + bo[n], f32 out
// ---------------------------------------------------------------------------
__global__ void __launch_bounds__(256)
out_gemm(const _Float16* __restrict__ A, const _Float16* __restrict__ W,
         const float* __restrict__ bias, float* __restrict__ out) {
  const int lane = threadIdx.x & 31;
  const int w    = threadIdx.x >> 5;
  const int r    = lane & 15;
  const int hi   = lane >> 4;
  const int m0   = blockIdx.x * 128 + (w & 3) * 32;
  const int n0   = blockIdx.y * 64 + (w >> 2) * 32;

  v8f acc[2][2] = {};

  const _Float16* arow0 = A + (size_t)(m0 + r) * (Hc * DVc);
  const _Float16* arow1 = A + (size_t)(m0 + 16 + r) * (Hc * DVc);
  const _Float16* wrow0 = W + (size_t)(n0 + r) * (Hc * DVc);
  const _Float16* wrow1 = W + (size_t)(n0 + 16 + r) * (Hc * DVc);

  for (int k0 = 0; k0 < Hc * DVc; k0 += 32) {
    __builtin_prefetch(arow0 + k0 + 256, 0, 1);
    __builtin_prefetch(wrow0 + k0 + 256, 0, 1);
    v16h a0 = frag_f16(arow0 + k0, hi);
    v16h a1 = frag_f16(arow1 + k0, hi);
    v16h b0 = frag_f16(wrow0 + k0, hi);
    v16h b1 = frag_f16(wrow1 + k0, hi);
    acc[0][0] = WMMA_F16(a0, b0, acc[0][0]);
    acc[0][1] = WMMA_F16(a0, b1, acc[0][1]);
    acc[1][0] = WMMA_F16(a1, b0, acc[1][0]);
    acc[1][1] = WMMA_F16(a1, b1, acc[1][1]);
  }

#pragma unroll
  for (int ti = 0; ti < 2; ++ti) {
#pragma unroll
    for (int tj = 0; tj < 2; ++tj) {
      const int n = n0 + tj * 16 + r;
      const float bb = bias[n];
#pragma unroll
      for (int v = 0; v < 8; ++v) {
        const int m = m0 + ti * 16 + v + 8 * hi;
        out[(size_t)m * OUTc + n] = acc[ti][tj][v] + bb;
      }
    }
  }
}

// ---------------------------------------------------------------------------
extern "C" void kernel_launch(void* const* d_in, const int* in_sizes, int n_in,
                              void* d_out, int out_size, void* d_ws, size_t ws_size,
                              hipStream_t stream) {
  const float* query   = (const float*)d_in[0];
  const float* key     = (const float*)d_in[1];
  const float* value   = (const float*)d_in[2];
  const float* pos_emb = (const float*)d_in[3];
  const float* Wq = (const float*)d_in[4];
  const float* bq = (const float*)d_in[5];
  const float* Wk = (const float*)d_in[6];
  const float* bk = (const float*)d_in[7];
  const float* Wv = (const float*)d_in[8];
  const float* bv = (const float*)d_in[9];
  const float* Wp = (const float*)d_in[10];
  const float* bp = (const float*)d_in[11];
  const float* Wo = (const float*)d_in[12];
  const float* bo = (const float*)d_in[13];
  const float* u_bias = (const float*)d_in[14];
  const float* v_bias = (const float*)d_in[15];
  float* out = (float*)d_out;

  // ---- workspace carve-up (all f16), ~47 MB total ---------------------------
  char* ws = (char*)d_ws;
  const size_t act_elems = (size_t)Bc * Tc * Fc;     // 2M
  const size_t w_elems   = (size_t)Fc * Fc;          // 1M
  const size_t qkv_bytes = (size_t)Bc * Hc * Tc * DKc * sizeof(_Float16); // 4 MB

  _Float16* qu = (_Float16*)ws;  ws += qkv_bytes;
  _Float16* qv = (_Float16*)ws;  ws += qkv_bytes;
  _Float16* kf = (_Float16*)ws;  ws += qkv_bytes;
  _Float16* vT = (_Float16*)ws;  ws += qkv_bytes;
  _Float16* pf = (_Float16*)ws;  ws += (size_t)Hc * Tc * DKc * sizeof(_Float16);
  _Float16* xb = (_Float16*)ws;  ws += (size_t)Bc * Tc * Hc * DVc * sizeof(_Float16);

  _Float16* q16 = (_Float16*)ws; ws += act_elems * 2;
  _Float16* k16 = (_Float16*)ws; ws += act_elems * 2;
  _Float16* v16 = (_Float16*)ws; ws += act_elems * 2;
  _Float16* p16 = (_Float16*)ws; ws += (size_t)Tc * Fc * 2;
  _Float16* wq16 = (_Float16*)ws; ws += w_elems * 2;
  _Float16* wk16 = (_Float16*)ws; ws += w_elems * 2;
  _Float16* wv16 = (_Float16*)ws; ws += w_elems * 2;
  _Float16* wp16 = (_Float16*)ws; ws += w_elems * 2;
  _Float16* wo16 = (_Float16*)ws; ws += w_elems * 2;

  const dim3 blk(256);

  // ---- one-shot f32 -> f16 conversions --------------------------------------
  auto cvt = [&](const float* src, _Float16* dst, size_t n) {
    cvt_kernel<<<dim3((unsigned)(n / (256 * 8))), blk, 0, stream>>>(src, dst, (int)n);
  };
  cvt(query,   q16,  act_elems);
  cvt(key,     k16,  act_elems);
  cvt(value,   v16,  act_elems);
  cvt(pos_emb, p16,  (size_t)Tc * Fc);
  cvt(Wq, wq16, w_elems);
  cvt(Wk, wk16, w_elems);
  cvt(Wv, wv16, w_elems);
  cvt(Wp, wp16, w_elems);
  cvt(Wo, wo16, w_elems);

  // ---- projections (pure-f16 WMMA GEMMs) ------------------------------------
  proj_kernel<<<dim3(16, 16), blk, 0, stream>>>(q16, wq16, bq, 0, qu, qv, u_bias, v_bias);
  proj_kernel<<<dim3(16, 16), blk, 0, stream>>>(k16, wk16, bk, 1, kf, nullptr, nullptr, nullptr);
  proj_kernel<<<dim3(16, 16), blk, 0, stream>>>(v16, wv16, bv, 2, vT, nullptr, nullptr, nullptr);
  proj_kernel<<<dim3(8, 16), blk, 0, stream>>>(p16, wp16, bp, 3, pf, nullptr, nullptr, nullptr);

  // ---- attention (197632 B dynamic LDS: within CDNA5's 320 KB/WG limit) -----
  const int smem_bytes = 131072 + 65536 + 1024;
  (void)hipFuncSetAttribute((const void*)attn_kernel,
                            hipFuncAttributeMaxDynamicSharedMemorySize, smem_bytes);
  attn_kernel<<<dim3(Bc * Hc * (Tc / TS)), blk, smem_bytes, stream>>>(qu, qv, kf, vT, pf, xb);

  // ---- output projection ----------------------------------------------------
  out_gemm<<<dim3(16, 16), blk, 0, stream>>>(xb, wo16, bo, out);
}
